// StableFeatureExtractorGNN_60559038873712
// MI455X (gfx1250) — matrix operand order repaired and checked by reading, not compiled, not run on hardware
//
#include <hip/hip_runtime.h>
#include <math.h>

typedef __attribute__((ext_vector_type(16))) __bf16 v16bf;
typedef __attribute__((ext_vector_type(8)))  float  v8f;

#define DEV __device__ __forceinline__

DEV unsigned short f32_to_bf16(float f) {
  unsigned int u = __float_as_uint(f);
  u += 0x7FFFu + ((u >> 16) & 1u);      // round-to-nearest-even
  return (unsigned short)(u >> 16);
}

// ---------------------------------------------------------------- elementwise
__global__ void k_zero(float* __restrict__ p, long n) {
  long i = (long)blockIdx.x * blockDim.x + threadIdx.x;
  if (i < n) p[i] = 0.f;
}
__global__ void k_copy(float* __restrict__ d, const float* __restrict__ s, long n) {
  long i = (long)blockIdx.x * blockDim.x + threadIdx.x;
  if (i < n) d[i] = s[i];
}
__global__ void k_to_bf16(unsigned short* __restrict__ d, const float* __restrict__ s, long n) {
  long i = (long)blockIdx.x * blockDim.x + threadIdx.x;
  if (i < n) d[i] = f32_to_bf16(s[i]);
}
__global__ void k_elu(float* __restrict__ p, long n) {
  long i = (long)blockIdx.x * blockDim.x + threadIdx.x;
  if (i < n) { float v = p[i]; p[i] = v > 0.f ? v : (__expf(v) - 1.f); }
}

// ---------------------------------------------------------------- adjacency build
__global__ void k_scatter_edges(float* __restrict__ A, const int* __restrict__ ei, int E, int n) {
  int e = blockIdx.x * blockDim.x + threadIdx.x;
  if (e < E) atomicAdd(&A[(long)ei[E + e] * n + ei[e]], 1.f);   // A[dst, src] += 1
}

// ---------------------------------------------------------------- GCN normalization
__global__ void k_rowsum_dinv(const float* __restrict__ A, float* __restrict__ dinv, int n) {
  __shared__ float sm[256];
  int r = blockIdx.x, t = threadIdx.x;
  float s = 0.f;
  for (int j = t; j < n; j += blockDim.x) s += A[(long)r * n + j];
  sm[t] = s; __syncthreads();
  for (int w = 128; w > 0; w >>= 1) { if (t < w) sm[t] += sm[t + w]; __syncthreads(); }
  if (t == 0) {
    float d = A[(long)r * n + r];
    float tot = sm[0] + (d == 0.f ? 2.f : 0.f);   // improved=True fill=2 on empty diag
    dinv[r] = rsqrtf(tot);
  }
}
__global__ void k_gcn_norm_bf16(unsigned short* __restrict__ Ab, const float* __restrict__ A,
                                const float* __restrict__ dinv, int n) {
  long i = (long)blockIdx.x * blockDim.x + threadIdx.x;
  if (i >= (long)n * n) return;
  int r = (int)(i / n), c = (int)(i % n);
  float v = A[i];
  if (r == c && v == 0.f) v = 2.f;
  Ab[i] = f32_to_bf16(v * dinv[r] * dinv[c]);
}
__global__ void k_aug_prep(unsigned short* __restrict__ Ab, const float* __restrict__ A, int n) {
  long i = (long)blockIdx.x * blockDim.x + threadIdx.x;
  if (i >= (long)n * n) return;
  int r = (int)(i / n), c = (int)(i % n);
  Ab[i] = f32_to_bf16(A[i] + (r == c ? 1.f : 0.f));
}
__global__ void k_zero_diag(float* __restrict__ A, int n) {
  int i = blockIdx.x * blockDim.x + threadIdx.x;
  if (i < n) A[(long)i * n + i] = 0.f;
}

// ---------------------------------------------------------------- first XW (IN=3)
__global__ void k_xw3(unsigned short* __restrict__ ob, const float* __restrict__ x,
                      const float* __restrict__ W, int n) {
  long i = (long)blockIdx.x * blockDim.x + threadIdx.x;
  if (i >= (long)n * 64) return;
  int r = (int)(i >> 6), j = (int)(i & 63);
  float s = x[r * 3] * W[j] + x[r * 3 + 1] * W[64 + j] + x[r * 3 + 2] * W[128 + j];
  ob[i] = f32_to_bf16(s);
}

// ---------------------------------------------------------------- generic WMMA GEMM
// C[M,N] = act(A@B + bias); one 16x64 C tile per wave (A fragment reused 4x).
__global__ void k_gemm(const unsigned short* __restrict__ A, const unsigned short* __restrict__ B,
                       const float* __restrict__ bias, float* __restrict__ C,
                       int M, int N, int K, int lda, int ldb, int ldc, int relu) {
  int lane = threadIdx.x & 31, wave = threadIdx.x >> 5;
  int tilesN = N >> 6;
  long tiles = (long)(M >> 4) * tilesN;
  long wid = (long)blockIdx.x * 8 + wave;
  if (wid >= tiles) return;
  int tM = (int)(wid / tilesN), tN = (int)(wid % tilesN);
  int m = lane & 15, hi = lane >> 4;

  v8f acc[4];
#pragma unroll
  for (int t = 0; t < 4; ++t) acc[t] = (v8f){0.f,0.f,0.f,0.f,0.f,0.f,0.f,0.f};

  const unsigned short* Arow = A + (long)(tM * 16 + m) * lda;
  for (int k0 = 0; k0 < K; k0 += 32) {
    union { v16bf v; unsigned int u[8]; } af;
#pragma unroll
    for (int v = 0; v < 8; ++v) {
      int kb = (v < 4 ? 2 * v : 16 + 2 * (v - 4)) + hi * 8;
      af.u[v] = *(const unsigned int*)(Arow + k0 + kb);
    }
    const unsigned short* Bk = B + (long)k0 * ldb + tN * 64 + m;
#pragma unroll
    for (int t = 0; t < 4; ++t) {
      union { v16bf v; unsigned int u[8]; } bf;
#pragma unroll
      for (int v = 0; v < 8; ++v) {
        int kb = (v < 4 ? 2 * v : 16 + 2 * (v - 4)) + hi * 8;
        bf.u[v] = (unsigned int)Bk[(long)kb * ldb + t * 16] | ((unsigned int)Bk[(long)(kb + 1) * ldb + t * 16] << 16);
      }
      acc[t] = __builtin_amdgcn_wmma_f32_16x16x32_bf16(false, af.v, false, bf.v,
                                                       (short)0, acc[t], false, false);
    }
  }
#pragma unroll
  for (int t = 0; t < 4; ++t) {
    int col = tN * 64 + t * 16 + m;
    float bv = bias ? bias[col] : 0.f;
#pragma unroll
    for (int r = 0; r < 8; ++r) {
      int row = tM * 16 + r + 8 * hi;
      float v = acc[t][r] + bv;
      if (relu) v = v > 0.f ? v : 0.f;
      C[(long)row * ldc + col] = v;
    }
  }
}

// ---------------------------------------------------------------- LDS-staged WMMA GEMM
// 128xBN block per 256-thread workgroup; A(128x32)/B(32xBN) tiles double-buffered in
// LDS via gfx1250 async global->LDS copies (ASYNCcnt), 8 waves each compute 16xBN.
// Requires M%128==0, N%BN==0, K%32==0, lda/ldb multiples of 8.
#define ASTR 40            // padded LDS stride for A tile (elements), 80B = 5*16B
template<int BN>
__global__ void k_gemm_lds(const unsigned short* __restrict__ A,
                           const unsigned short* __restrict__ B,
                           const float* __restrict__ bias, float* __restrict__ C,
                           int M, int N, int K, int lda, int ldb, int ldc, int relu) {
  constexpr int BSTR = BN + 8;     // padded LDS stride for B tile, multiple of 8 elem
  __shared__ unsigned short As[2][128 * ASTR];
  __shared__ unsigned short Bs[2][32 * BSTR];

  int tid = threadIdx.x, lane = tid & 31, wave = tid >> 5;
  int nbn = N / BN;
  int bM = blockIdx.x / nbn, bN = blockIdx.x % nbn;
  const unsigned short* Ag = A + (long)bM * 128 * lda;
  const unsigned short* Bg = B + bN * BN;

  auto issue = [&](int buf, int k0) {
    unsigned int aBase = (unsigned int)(size_t)&As[buf][0];
    unsigned int bBase = (unsigned int)(size_t)&Bs[buf][0];
    // A tile: 128x32 = 512 chunks of 8 elements -> 2 per thread
#pragma unroll
    for (int c = 0; c < 2; ++c) {
      int ch = tid * 2 + c;
      int row = ch >> 2, col = (ch & 3) * 8;
      const unsigned short* ga = Ag + (long)row * lda + k0 + col;
      unsigned int la = aBase + (unsigned int)(row * ASTR + col) * 2u;
      asm volatile("global_load_async_to_lds_b128 %0, %1, off"
                   :: "v"(la), "v"(ga) : "memory");
    }
    // B tile: 32xBN -> BN*4 chunks of 8 elements -> BN/64 per thread
#pragma unroll
    for (int c = 0; c < BN / 64; ++c) {
      int ch = tid + c * 256;
      int row = ch / (BN / 8), col = (ch % (BN / 8)) * 8;
      const unsigned short* gb = Bg + (long)(k0 + row) * ldb + col;
      unsigned int lb = bBase + (unsigned int)(row * BSTR + col) * 2u;
      asm volatile("global_load_async_to_lds_b128 %0, %1, off"
                   :: "v"(lb), "v"(gb) : "memory");
    }
  };

  int m = lane & 15, hi = lane >> 4;
  v8f acc[BN / 16];
#pragma unroll
  for (int t = 0; t < BN / 16; ++t) acc[t] = (v8f){0.f,0.f,0.f,0.f,0.f,0.f,0.f,0.f};

  int nk = K >> 5;
  issue(0, 0);
  for (int i = 0; i < nk; ++i) {
    int buf = i & 1;
    if (i + 1 < nk) {
      issue(buf ^ 1, (i + 1) << 5);
      // async loads complete in order: waiting <= next-buffer op count
      // guarantees the current buffer's copies have landed in LDS.
      if constexpr (BN == 128) asm volatile("s_wait_asynccnt 4" ::: "memory");
      else                     asm volatile("s_wait_asynccnt 3" ::: "memory");
    } else {
      asm volatile("s_wait_asynccnt 0" ::: "memory");
    }
    __syncthreads();

    union { v16bf v; unsigned int u[8]; } af;
    const unsigned short* Asr = &As[buf][(wave * 16 + m) * ASTR];
#pragma unroll
    for (int v = 0; v < 8; ++v) {
      int kb = (v < 4 ? 2 * v : 16 + 2 * (v - 4)) + hi * 8;
      af.u[v] = *(const unsigned int*)(Asr + kb);
    }
    const unsigned short* Bsr = &Bs[buf][m];
#pragma unroll
    for (int t = 0; t < BN / 16; ++t) {
      union { v16bf v; unsigned int u[8]; } bf;
#pragma unroll
      for (int v = 0; v < 8; ++v) {
        int kb = (v < 4 ? 2 * v : 16 + 2 * (v - 4)) + hi * 8;
        bf.u[v] = (unsigned int)Bsr[kb * BSTR + t * 16] | ((unsigned int)Bsr[(kb + 1) * BSTR + t * 16] << 16);
      }
      acc[t] = __builtin_amdgcn_wmma_f32_16x16x32_bf16(false, af.v, false, bf.v,
                                                       (short)0, acc[t], false, false);
    }
    __syncthreads();
  }

#pragma unroll
  for (int t = 0; t < BN / 16; ++t) {
    int col = bN * BN + t * 16 + m;
    float bv = bias ? bias[col] : 0.f;
#pragma unroll
    for (int r = 0; r < 8; ++r) {
      int row = bM * 128 + wave * 16 + r + 8 * hi;
      float v = acc[t][r] + bv;
      if (relu) v = v > 0.f ? v : 0.f;
      C[(long)row * ldc + col] = v;
    }
  }
}

// ---------------------------------------------------------------- TopK pooling
__global__ void k_pnorm_inv(float* __restrict__ inv, const float* __restrict__ p, int f) {
  __shared__ float sm[64];
  int t = threadIdx.x;
  sm[t] = (t < f) ? p[t] * p[t] : 0.f; __syncthreads();
  for (int w = 32; w > 0; w >>= 1) { if (t < w) sm[t] += sm[t + w]; __syncthreads(); }
  if (t == 0) inv[0] = rsqrtf(sm[0]);
}
__global__ void k_pool_score(float* __restrict__ sc, const float* __restrict__ x,
                             const float* __restrict__ p, const float* __restrict__ pinv,
                             int n, int f) {
  int i = blockIdx.x * blockDim.x + threadIdx.x;
  if (i >= n) return;
  float s = 0.f;
  for (int c = 0; c < f; ++c) s += x[(long)i * f + c] * p[c];
  sc[i] = tanhf(s * pinv[0]);
}
// stable descending rank (matches jax.lax.top_k ordering)
__global__ void k_topk(int* __restrict__ perm, const float* __restrict__ sc, int n, int k) {
  int i = blockIdx.x * blockDim.x + threadIdx.x;
  if (i >= n) return;
  float si = sc[i];
  int cnt = 0;
  for (int j = 0; j < n; ++j) {
    float sj = sc[j];
    cnt += (sj > si) || (sj == si && j < i);
  }
  if (cnt < k) perm[cnt] = i;
}
__global__ void k_gather_A(float* __restrict__ An, const float* __restrict__ Aaug,
                           const int* __restrict__ perm, int nold, int k) {
  long idx = (long)blockIdx.x * blockDim.x + threadIdx.x;
  if (idx >= (long)k * k) return;
  int i = (int)(idx / k), j = (int)(idx % k);
  An[idx] = Aaug[(long)perm[i] * nold + perm[j]];
}
__global__ void k_gather_x(float* __restrict__ h, const float* __restrict__ x,
                           const int* __restrict__ perm, const float* __restrict__ sc,
                           int k, int f) {
  long idx = (long)blockIdx.x * blockDim.x + threadIdx.x;
  if (idx >= (long)k * f) return;
  int i = (int)(idx / f), c = (int)(idx % f);
  int p = perm[i];
  h[idx] = x[(long)p * f + c] * sc[p];
}
__global__ void k_scatter_add(float* __restrict__ dst, const float* __restrict__ src,
                              const int* __restrict__ perm, int k, int f) {
  long idx = (long)blockIdx.x * blockDim.x + threadIdx.x;
  if (idx >= (long)k * f) return;
  int i = (int)(idx / f), c = (int)(idx % f);
  dst[(long)perm[i] * f + c] += src[idx];
}

// ---------------------------------------------------------------- LayerNorm (f=128)
__global__ void k_layernorm(float* __restrict__ u, const float* __restrict__ h,
                            const float* __restrict__ g, const float* __restrict__ b, int f) {
  __shared__ float sm[128];
  int r = blockIdx.x, t = threadIdx.x;
  float v = h[(long)r * f + t];
  sm[t] = v; __syncthreads();
  for (int w = 64; w > 0; w >>= 1) { if (t < w) sm[t] += sm[t + w]; __syncthreads(); }
  float mu = sm[0] / f; __syncthreads();
  float d = v - mu;
  sm[t] = d * d; __syncthreads();
  for (int w = 64; w > 0; w >>= 1) { if (t < w) sm[t] += sm[t + w]; __syncthreads(); }
  float var = sm[0] / f;
  u[(long)r * f + t] = d * rsqrtf(var + 1e-6f) * g[t] + b[t];
}

// ---------------------------------------------------------------- GAT
__global__ void k_ed_es(float* __restrict__ es, float* __restrict__ ed,
                        const float* __restrict__ H, const float* __restrict__ asrc,
                        const float* __restrict__ adst, int n, int heads, int c) {
  int idx = blockIdx.x * blockDim.x + threadIdx.x;
  if (idx >= n * heads) return;
  int s = idx / heads, h = idx % heads;
  const float* row = H + (long)s * heads * c + h * c;
  float a = 0.f, d = 0.f;
  for (int cc = 0; cc < c; ++cc) { a += row[cc] * asrc[h * c + cc]; d += row[cc] * adst[h * c + cc]; }
  es[idx] = a; ed[idx] = d;
}
// per-(dst,head) masked softmax statistics: row max + sum(exp)
__global__ void k_gat_stats(float* __restrict__ rmax, float* __restrict__ rsum,
                            const float* __restrict__ A, const float* __restrict__ es,
                            const float* __restrict__ ed, int n, int heads) {
  __shared__ float sm[256];
  int d = blockIdx.x / heads, h = blockIdx.x % heads, t = threadIdx.x;
  float edv = ed[d * heads + h];
  const float* Ad = A + (long)d * n;
  float mx = -3.4e38f;
  for (int s = t; s < n; s += blockDim.x)
    if (Ad[s] > 0.f || s == d) {
      float lg = edv + es[s * heads + h];
      lg = lg > 0.f ? lg : 0.2f * lg;
      mx = fmaxf(mx, lg);
    }
  sm[t] = mx; __syncthreads();
  for (int w = 128; w > 0; w >>= 1) { if (t < w) sm[t] = fmaxf(sm[t], sm[t + w]); __syncthreads(); }
  mx = sm[0]; __syncthreads();
  float sum = 0.f;
  for (int s = t; s < n; s += blockDim.x)
    if (Ad[s] > 0.f || s == d) {
      float lg = edv + es[s * heads + h];
      lg = lg > 0.f ? lg : 0.2f * lg;
      sum += __expf(lg - mx);
    }
  sm[t] = sum; __syncthreads();
  for (int w = 128; w > 0; w >>= 1) { if (t < w) sm[t] += sm[t + w]; __syncthreads(); }
  if (t == 0) { rmax[blockIdx.x] = mx; rsum[blockIdx.x] = sm[0]; }
}
// out[d, h*c + cc] = sum_s alpha[d,s,h] * Hb[s, h*c+cc]  (+bias, +optional residual)
// alpha built in registers directly in the WMMA A-fragment layout.
__global__ void k_gat_out(float* __restrict__ out, const float* __restrict__ A,
                          const float* __restrict__ es, const float* __restrict__ ed,
                          const float* __restrict__ rmax, const float* __restrict__ rsum,
                          const unsigned short* __restrict__ Hb, const float* __restrict__ bias,
                          const float* __restrict__ addsrc, int n, int heads, int c) {
  int lane = threadIdx.x & 31, wave = threadIdx.x >> 5;
  int cTiles = c >> 4, ldh = heads * c;
  long wid = (long)blockIdx.x * 8 + wave;
  long total = (long)(n >> 4) * heads * cTiles;
  if (wid >= total) return;
  int tC = (int)(wid % cTiles); wid /= cTiles;
  int h = (int)(wid % heads);   wid /= heads;
  int tD = (int)wid;
  int m = lane & 15, hi = lane >> 4;
  int d = tD * 16 + m;
  float edv = ed[d * heads + h];
  float mx = rmax[d * heads + h], inv = 1.f / rsum[d * heads + h];
  const float* Ad = A + (long)d * n;

  v8f acc = {0.f, 0.f, 0.f, 0.f, 0.f, 0.f, 0.f, 0.f};
  for (int k0 = 0; k0 < n; k0 += 32) {
    union { v16bf v; unsigned short s16[16]; } af;
    union { v16bf v; unsigned int u[8]; } bf;
#pragma unroll
    for (int v = 0; v < 8; ++v) {
      int kb = (v < 4 ? 2 * v : 16 + 2 * (v - 4)) + hi * 8;
#pragma unroll
      for (int t = 0; t < 2; ++t) {
        int s = k0 + kb + t;
        float al = 0.f;
        if (Ad[s] > 0.f || s == d) {
          float lg = edv + es[s * heads + h];
          lg = lg > 0.f ? lg : 0.2f * lg;
          al = __expf(lg - mx) * inv;
        }
        af.s16[2 * v + t] = f32_to_bf16(al);
      }
    }
    const unsigned short* Bk = Hb + (long)k0 * ldh + h * c + tC * 16 + m;
#pragma unroll
    for (int v = 0; v < 8; ++v) {
      int kb = (v < 4 ? 2 * v : 16 + 2 * (v - 4)) + hi * 8;
      bf.u[v] = (unsigned int)Bk[(long)kb * ldh] | ((unsigned int)Bk[(long)(kb + 1) * ldh] << 16);
    }
    acc = __builtin_amdgcn_wmma_f32_16x16x32_bf16(false, af.v, false, bf.v,
                                                  (short)0, acc, false, false);
  }
  int co = h * c + tC * 16 + m;
  float bv = bias[co];
#pragma unroll
  for (int r = 0; r < 8; ++r) {
    int row = tD * 16 + r + 8 * hi;
    float v = acc[r] + bv;
    if (addsrc) v += addsrc[(long)row * ldh + co];
    out[(long)row * ldh + co] = v;
  }
}

// ================================================================ host side
static inline int cdiv(long a, int b) { return (int)((a + b - 1) / b); }

static void gemm(hipStream_t st, const unsigned short* A, const unsigned short* B,
                 const float* bias, float* C, int M, int N, int K,
                 int lda, int ldb, int ldc, int relu) {
  long tiles = (long)(M >> 4) * (N >> 6);
  k_gemm<<<cdiv(tiles, 8), 256, 0, st>>>(A, B, bias, C, M, N, K, lda, ldb, ldc, relu);
}
static void gemm_lds(hipStream_t st, const unsigned short* A, const unsigned short* B,
                     const float* bias, float* C, int M, int N, int K,
                     int lda, int ldb, int ldc, int relu) {
  if (N % 128 == 0)
    k_gemm_lds<128><<<(M / 128) * (N / 128), 256, 0, st>>>(A, B, bias, C, M, N, K, lda, ldb, ldc, relu);
  else
    k_gemm_lds<64><<<(M / 128) * (N / 64), 256, 0, st>>>(A, B, bias, C, M, N, K, lda, ldb, ldc, relu);
}

extern "C" void kernel_launch(void* const* d_in, const int* in_sizes, int n_in,
                              void* d_out, int out_size, void* d_ws, size_t ws_size,
                              hipStream_t stream) {
  const int N0 = 4096, N1 = 2048, N2 = 1024, N3 = 512, F = 64, OC = 128, HD = 4, G1 = HD * OC;
  const int E = in_sizes[1] / 2;

  const float* x        = (const float*)d_in[0];
  const int*   ei       = (const int*)d_in[1];
  const float* unet_w0  = (const float*)d_in[2];  const float* unet_b0 = (const float*)d_in[3];
  const float* unet_w1  = (const float*)d_in[4];  const float* unet_b1 = (const float*)d_in[5];
  const float* unet_w2  = (const float*)d_in[6];  const float* unet_b2 = (const float*)d_in[7];
  const float* unet_w3  = (const float*)d_in[8];  const float* unet_b3 = (const float*)d_in[9];
  const float* pool_p1  = (const float*)d_in[10];
  const float* pool_p2  = (const float*)d_in[11];
  const float* pool_p3  = (const float*)d_in[12];
  const float* up_w0    = (const float*)d_in[13]; const float* up_b0 = (const float*)d_in[14];
  const float* up_w1    = (const float*)d_in[15]; const float* up_b1 = (const float*)d_in[16];
  const float* up_w2    = (const float*)d_in[17]; const float* up_b2 = (const float*)d_in[18];
  const float* ln_g     = (const float*)d_in[19]; const float* ln_b  = (const float*)d_in[20];
  const float* res_w    = (const float*)d_in[21]; const float* res_b = (const float*)d_in[22];
  const float* gat1_w   = (const float*)d_in[23];
  const float* gat1_as  = (const float*)d_in[24]; const float* gat1_ad = (const float*)d_in[25];
  const float* gat1_b   = (const float*)d_in[26];
  const float* gat2_w   = (const float*)d_in[27];
  const float* gat2_as  = (const float*)d_in[28]; const float* gat2_ad = (const float*)d_in[29];
  const float* gat2_b   = (const float*)d_in[30];
  float* outp = (float*)d_out;

  // ---- workspace carve-out ----
  char* w = (char*)d_ws; size_t off = 0;
  auto alloc = [&](size_t bytes) -> void* {
    void* p = w + off; off += (bytes + 255) & ~(size_t)255; return p;
  };
  float* A0   = (float*)alloc((size_t)N0 * N0 * 4);
  float* Aaug = (float*)alloc((size_t)N0 * N0 * 4);
  unsigned short* Abf = (unsigned short*)alloc((size_t)N0 * N0 * 2);
  float* A1 = (float*)alloc((size_t)N1 * N1 * 4);
  float* A2 = (float*)alloc((size_t)N2 * N2 * 4);
  float* A3 = (float*)alloc((size_t)N3 * N3 * 4);
  unsigned short* xb  = (unsigned short*)alloc((size_t)N0 * F * 2);
  unsigned short* wb  = (unsigned short*)alloc((size_t)G1 * OC * 2);
  float*          xwf = (float*)alloc((size_t)N0 * OC * 4);
  unsigned short* xwb = (unsigned short*)alloc((size_t)N0 * OC * 2);
  float* x0 = (float*)alloc((size_t)N0 * F * 4);
  float* h1 = (float*)alloc((size_t)N1 * F * 4);
  float* x1 = (float*)alloc((size_t)N1 * F * 4);
  float* h2 = (float*)alloc((size_t)N2 * F * 4);
  float* x2 = (float*)alloc((size_t)N2 * F * 4);
  float* h3 = (float*)alloc((size_t)N3 * F * 4);
  float* x3 = (float*)alloc((size_t)N3 * F * 4);
  float* hup2 = (float*)alloc((size_t)N2 * F * 4);
  float* hu2  = (float*)alloc((size_t)N2 * F * 4);
  float* hup1 = (float*)alloc((size_t)N1 * F * 4);
  float* hu1  = (float*)alloc((size_t)N1 * F * 4);
  float* hup0 = (float*)alloc((size_t)N0 * F * 4);
  float* hfin = (float*)alloc((size_t)N0 * OC * 4);
  float* uu   = (float*)alloc((size_t)N0 * OC * 4);
  unsigned short* ub = (unsigned short*)alloc((size_t)N0 * OC * 2);
  float* resid = (float*)alloc((size_t)N0 * OC * 4);
  float* hfeat = (float*)alloc((size_t)N0 * G1 * 4);
  unsigned short* hfb = (unsigned short*)alloc((size_t)N0 * G1 * 2);
  float* hfeat2 = (float*)alloc((size_t)N0 * OC * 4);
  float* dinv0 = (float*)alloc(N0 * 4); float* dinv1 = (float*)alloc(N1 * 4);
  float* dinv2 = (float*)alloc(N2 * 4); float* dinv3 = (float*)alloc(N3 * 4);
  float* sc1 = (float*)alloc(N0 * 4); float* sc2 = (float*)alloc(N1 * 4); float* sc3 = (float*)alloc(N2 * 4);
  int* perm1 = (int*)alloc(N1 * 4); int* perm2 = (int*)alloc(N2 * 4); int* perm3 = (int*)alloc(N3 * 4);
  float* es1 = (float*)alloc(N0 * HD * 4); float* ed1 = (float*)alloc(N0 * HD * 4);
  float* rmx1 = (float*)alloc(N0 * HD * 4); float* rsm1 = (float*)alloc(N0 * HD * 4);
  float* es2 = (float*)alloc(N0 * 4); float* ed2 = (float*)alloc(N0 * 4);
  float* rmx2 = (float*)alloc(N0 * 4); float* rsm2 = (float*)alloc(N0 * 4);
  float* pinv = (float*)alloc(256);
  (void)ws_size; (void)n_in; (void)out_size;

  const int BS = 256;

  // ---- dense weighted adjacency ----
  k_zero<<<cdiv((long)N0 * N0, BS), BS, 0, stream>>>(A0, (long)N0 * N0);
  k_scatter_edges<<<cdiv(E, BS), BS, 0, stream>>>(A0, ei, E, N0);

  // generic GCN step: out = act(norm(A) @ (X@W) + b)
  auto gcn = [&](const float* A, int n, const float* dinv, const float* X, int fin,
                 const float* W, const float* b, float* out, int fout, int relu) {
    k_to_bf16<<<cdiv((long)fin * fout, BS), BS, 0, stream>>>(wb, W, (long)fin * fout);
    k_to_bf16<<<cdiv((long)n * fin, BS), BS, 0, stream>>>(xb, X, (long)n * fin);
    gemm(stream, xb, wb, nullptr, xwf, n, fout, fin, fin, fout, fout, 0);
    k_to_bf16<<<cdiv((long)n * fout, BS), BS, 0, stream>>>(xwb, xwf, (long)n * fout);
    k_gcn_norm_bf16<<<cdiv((long)n * n, BS), BS, 0, stream>>>(Abf, A, dinv, n);
    gemm_lds(stream, Abf, xwb, b, out, n, fout, n, n, fout, fout, relu);
  };
  // augment: Aaug = (A+I)@(A+I), zero diagonal
  auto augment = [&](const float* A, int n) {
    k_aug_prep<<<cdiv((long)n * n, BS), BS, 0, stream>>>(Abf, A, n);
    gemm_lds(stream, Abf, Abf, nullptr, Aaug, n, n, n, n, n, n, 0);
    k_zero_diag<<<cdiv(n, BS), BS, 0, stream>>>(Aaug, n);
  };
  // top-k pool: Aout = Aaug[perm][:,perm], hout = xin[perm]*score[perm]
  auto pool = [&](const float* xin, int n, int k, const float* p, float* sc, int* perm,
                  float* Aout, float* hout) {
    k_pnorm_inv<<<1, 64, 0, stream>>>(pinv, p, F);
    k_pool_score<<<cdiv(n, BS), BS, 0, stream>>>(sc, xin, p, pinv, n, F);
    k_topk<<<cdiv(n, BS), BS, 0, stream>>>(perm, sc, n, k);
    k_gather_A<<<cdiv((long)k * k, BS), BS, 0, stream>>>(Aout, Aaug, perm, n, k);
    k_gather_x<<<cdiv((long)k * F, BS), BS, 0, stream>>>(hout, xin, perm, sc, k, F);
  };

  // ---- GraphUNet down path ----
  k_rowsum_dinv<<<N0, 256, 0, stream>>>(A0, dinv0, N0);
  k_xw3<<<cdiv((long)N0 * F, BS), BS, 0, stream>>>(xwb, x, unet_w0, N0);
  k_gcn_norm_bf16<<<cdiv((long)N0 * N0, BS), BS, 0, stream>>>(Abf, A0, dinv0, N0);
  gemm_lds(stream, Abf, xwb, unet_b0, x0, N0, F, N0, N0, F, F, 1);

  augment(A0, N0);
  pool(x0, N0, N1, pool_p1, sc1, perm1, A1, h1);
  k_rowsum_dinv<<<N1, 256, 0, stream>>>(A1, dinv1, N1);
  gcn(A1, N1, dinv1, h1, F, unet_w1, unet_b1, x1, F, 1);

  augment(A1, N1);
  pool(x1, N1, N2, pool_p2, sc2, perm2, A2, h2);
  k_rowsum_dinv<<<N2, 256, 0, stream>>>(A2, dinv2, N2);
  gcn(A2, N2, dinv2, h2, F, unet_w2, unet_b2, x2, F, 1);

  augment(A2, N2);
  pool(x2, N2, N3, pool_p3, sc3, perm3, A3, h3);
  k_rowsum_dinv<<<N3, 256, 0, stream>>>(A3, dinv3, N3);
  gcn(A3, N3, dinv3, h3, F, unet_w3, unet_b3, x3, F, 1);

  // ---- up path (sum_res=True) ----
  k_copy<<<cdiv((long)N2 * F, BS), BS, 0, stream>>>(hup2, x2, (long)N2 * F);
  k_scatter_add<<<cdiv((long)N3 * F, BS), BS, 0, stream>>>(hup2, x3, perm3, N3, F);
  gcn(A2, N2, dinv2, hup2, F, up_w0, up_b0, hu2, F, 1);

  k_copy<<<cdiv((long)N1 * F, BS), BS, 0, stream>>>(hup1, x1, (long)N1 * F);
  k_scatter_add<<<cdiv((long)N2 * F, BS), BS, 0, stream>>>(hup1, hu2, perm2, N2, F);
  gcn(A1, N1, dinv1, hup1, F, up_w1, up_b1, hu1, F, 1);

  k_copy<<<cdiv((long)N0 * F, BS), BS, 0, stream>>>(hup0, x0, (long)N0 * F);
  k_scatter_add<<<cdiv((long)N1 * F, BS), BS, 0, stream>>>(hup0, hu1, perm1, N1, F);
  gcn(A0, N0, dinv0, hup0, F, up_w2, up_b2, hfin, OC, 0);   // last conv: no relu

  // ---- LayerNorm + residual projection ----
  k_layernorm<<<N0, OC, 0, stream>>>(uu, hfin, ln_g, ln_b, OC);
  k_to_bf16<<<cdiv((long)N0 * OC, BS), BS, 0, stream>>>(ub, uu, (long)N0 * OC);
  k_to_bf16<<<cdiv((long)OC * OC, BS), BS, 0, stream>>>(wb, res_w, (long)OC * OC);
  gemm(stream, ub, wb, res_b, resid, N0, OC, OC, OC, OC, OC, 0);

  // ---- GAT layer 1 (4 heads, concat) ----
  k_to_bf16<<<cdiv((long)OC * G1, BS), BS, 0, stream>>>(wb, gat1_w, (long)OC * G1);
  gemm(stream, ub, wb, nullptr, hfeat, N0, G1, OC, OC, G1, G1, 0);
  k_to_bf16<<<cdiv((long)N0 * G1, BS), BS, 0, stream>>>(hfb, hfeat, (long)N0 * G1);
  k_ed_es<<<cdiv(N0 * HD, BS), BS, 0, stream>>>(es1, ed1, hfeat, gat1_as, gat1_ad, N0, HD, OC);
  k_gat_stats<<<N0 * HD, 256, 0, stream>>>(rmx1, rsm1, A0, es1, ed1, N0, HD);
  {
    long tiles = (long)(N0 >> 4) * HD * (OC >> 4);
    k_gat_out<<<cdiv(tiles, 8), 256, 0, stream>>>(hfeat, A0, es1, ed1, rmx1, rsm1,
                                                  hfb, gat1_b, nullptr, N0, HD, OC);
  }
  k_elu<<<cdiv((long)N0 * G1, BS), BS, 0, stream>>>(hfeat, (long)N0 * G1);  // g = elu(out+b)

  // ---- GAT layer 2 (1 head) + residual -> output ----
  k_to_bf16<<<cdiv((long)N0 * G1, BS), BS, 0, stream>>>(hfb, hfeat, (long)N0 * G1);
  k_to_bf16<<<cdiv((long)G1 * OC, BS), BS, 0, stream>>>(wb, gat2_w, (long)G1 * OC);
  gemm(stream, hfb, wb, nullptr, hfeat2, N0, OC, G1, G1, OC, OC, 0);
  k_to_bf16<<<cdiv((long)N0 * OC, BS), BS, 0, stream>>>(xwb, hfeat2, (long)N0 * OC);
  k_ed_es<<<cdiv(N0, BS), BS, 0, stream>>>(es2, ed2, hfeat2, gat2_as, gat2_ad, N0, 1, OC);
  k_gat_stats<<<N0, 256, 0, stream>>>(rmx2, rsm2, A0, es2, ed2, N0, 1);
  {
    long tiles = (long)(N0 >> 4) * (OC >> 4);
    k_gat_out<<<cdiv(tiles, 8), 256, 0, stream>>>(outp, A0, es2, ed2, rmx2, rsm2,
                                                  xwb, gat2_b, resid, N0, 1, OC);
  }
}
